// Shared_Module_GCN_18399639896907
// MI455X (gfx1250) — matrix operand order repaired and checked by reading, not compiled, not run on hardware
//
#include <hip/hip_runtime.h>
#include <hip/hip_bf16.h>

typedef __attribute__((ext_vector_type(16))) __bf16 v16bf;
typedef __attribute__((ext_vector_type(8)))  float  v8f;
typedef __attribute__((ext_vector_type(4)))  float  v4f;

#define KNN_K   20
#define NPTS    2048
#define NBATCH  8
#define CATCH   512
#define EPSBN   1e-5f

// ---------------------------------------------------------------------------
// helpers
// ---------------------------------------------------------------------------
__device__ __forceinline__ void atomic_max_f32(float* addr, float v) {
  // monotone bit trick: works for mixed-sign floats
  if (v >= 0.0f) atomicMax((int*)addr, __float_as_int(v));
  else           atomicMin((unsigned int*)addr, __float_as_uint(v));
}

// A-matrix 16x32 bf16 fragment (ISA 7.12.2): lane holds row M=lane&15,
// K in {kb8..kb8+7} (V0..3) and {kb8+16..kb8+23} (V4..7), kb8=(lane>>4)*8.
__device__ __forceinline__ v16bf load_frag_a(const __bf16* rowbase, int half) {
  union { v16bf v; v4f f[2]; } u;
  u.f[0] = *(const v4f*)(rowbase + half * 8);
  u.f[1] = *(const v4f*)(rowbase + half * 8 + 16);
  return u.v;
}
// B-matrix 32x16 bf16 fragment: lane holds column N=lane&15,
// K in {kb8..kb8+7} and {kb8+16..kb8+23}, kb8=(lane>>4)*8.  rowbase = W row (o, c..)
__device__ __forceinline__ v16bf load_frag_b(const __bf16* rowbase, int half) {
  union { v16bf v; v4f f[2]; } u;
  u.f[0] = *(const v4f*)(rowbase + half * 8);
  u.f[1] = *(const v4f*)(rowbase + half * 8 + 16);
  return u.v;
}

// ---------------------------------------------------------------------------
// kNN: one thread per point, xyz staged in LDS, top-20 of -dist^2 (self incl.)
// ---------------------------------------------------------------------------
__global__ void knn_kernel(const float* __restrict__ x, int* __restrict__ nbr) {
  __shared__ float xs[NPTS * 3];
  const int b = blockIdx.y;
  for (int e = threadIdx.x; e < NPTS * 3; e += 256)
    xs[e] = x[(size_t)b * NPTS * 3 + e];
  __syncthreads();

  const int n = blockIdx.x * 256 + threadIdx.x;
  const float px = xs[n * 3], py = xs[n * 3 + 1], pz = xs[n * 3 + 2];

  float bd[KNN_K]; int bi[KNN_K];
#pragma unroll
  for (int t = 0; t < KNN_K; ++t) { bd[t] = -3.0e38f; bi[t] = 0; }

  for (int j = 0; j < NPTS; ++j) {
    float dx = xs[j * 3] - px, dy = xs[j * 3 + 1] - py, dz = xs[j * 3 + 2] - pz;
    float d = -(dx * dx + dy * dy + dz * dz);
    float mn = bd[0]; int pos = 0;
#pragma unroll
    for (int t = 1; t < KNN_K; ++t) if (bd[t] < mn) { mn = bd[t]; pos = t; }
    if (d > mn) {
#pragma unroll
      for (int t = 0; t < KNN_K; ++t) if (t == pos) { bd[t] = d; bi[t] = j; }
    }
  }
#pragma unroll
  for (int t = 0; t < KNN_K; ++t)
    nbr[((size_t)b * NPTS + n) * KNN_K + t] = bi[t];
}

// ---------------------------------------------------------------------------
// small utility kernels
// ---------------------------------------------------------------------------
__global__ void fill_kernel(float* p, float v, int n) {
  int t = blockIdx.x * 256 + threadIdx.x;
  if (t < n) p[t] = v;
}
__global__ void cvt_feat_kernel(const float* __restrict__ s, __bf16* __restrict__ d, int n) {
  int t = blockIdx.x * 256 + threadIdx.x;
  if (t < n) d[t] = (__bf16)s[t];
}
// pad weight rows [O][C2] f32 -> [O][KP] bf16, zeros past C2
__global__ void cvt_w_kernel(const float* __restrict__ w, __bf16* __restrict__ wp,
                             int O, int C2, int KP) {
  int t = blockIdx.x * 256 + threadIdx.x;
  if (t >= O * KP) return;
  int o = t / KP, c = t - o * KP;
  wp[t] = (c < C2) ? (__bf16)w[o * C2 + c] : (__bf16)0.0f;
}
// fbuf [bn][O] f32 -> fcat [bn][512] bf16 at channel offset
__global__ void cvt_cat_kernel(const float* __restrict__ fbuf, __bf16* __restrict__ fcat,
                               int O, int coff) {
  int t = blockIdx.x * 256 + threadIdx.x;
  if (t >= NBATCH * NPTS * O) return;
  int bn = t / O, c = t - bn * O;
  fcat[(size_t)bn * CATCH + coff + c] = (__bf16)fbuf[t];
}
__global__ void finalize_kernel(const float* __restrict__ sumb, const float* __restrict__ sqb,
                                const float* __restrict__ g, const float* __restrict__ bb,
                                float* __restrict__ scaleb, float* __restrict__ shiftb,
                                int O, float cnt) {
  int o = blockIdx.x * 256 + threadIdx.x;
  if (o >= O) return;
  float mean = sumb[o] / cnt;
  float var  = sqb[o] / cnt - mean * mean;
  float s = g[o] * rsqrtf(var + EPSBN);
  scaleb[o] = s;
  shiftb[o] = bb[o] - mean * s;
}
__global__ void init_out_kernel(float* out) {
  int t = blockIdx.x * 256 + threadIdx.x;
  if (t >= NBATCH * 2048) return;
  out[t] = ((t & 2047) < 1024) ? -2.0f : 0.0f;   // max slots / mean-sum slots
}
__global__ void finish_mean_kernel(float* out) {
  int t = blockIdx.x * 256 + threadIdx.x;
  if (t >= NBATCH * 1024) return;
  int b = t >> 10, o = t & 1023;
  out[b * 2048 + 1024 + o] *= (1.0f / NPTS);
}

// ---------------------------------------------------------------------------
// Edge-conv GEMM.  One wave per 16-row M tile of the (N*K_knn) x O output.
// PASS==1: accumulate per-channel sum/sumsq.  PASS==2: BN+tanh+max-over-k.
// ---------------------------------------------------------------------------
template<int PASS>
__global__ __launch_bounds__(32)
void edge_gemm_kernel(const __bf16* __restrict__ feats, int FS, int FOFF, int Cin,
                      const int* __restrict__ nbr, const __bf16* __restrict__ Wp,
                      int O, int KP,
                      float* __restrict__ sumb, float* __restrict__ sqb,
                      const float* __restrict__ scaleb, const float* __restrict__ shiftb,
                      float* __restrict__ fbuf) {
  extern __shared__ char smemraw[];
  int*    nrow  = (int*)smemraw;
  int*    nbrow = nrow + 16;
  __bf16* A     = (__bf16*)(smemraw + 128);

  const int lane  = threadIdx.x;
  const int nc    = lane & 15;
  const int half  = lane >> 4;
  const int mtile = blockIdx.x;
  const int b     = blockIdx.y;
  const int C2    = 2 * Cin;

  if (lane < 16) {
    int m = mtile * 16 + lane;
    int n = m / KNN_K;
    int k = m - n * KNN_K;
    nrow[lane]  = n;
    nbrow[lane] = nbr[((size_t)b * NPTS + n) * KNN_K + k];
  }
  __syncthreads();

  // build edge-feature tile: [16 rows][KP]  = [gathered-center | center | 0pad]
  const __bf16* fb = feats + (size_t)b * NPTS * FS + FOFF;
  for (int e = lane; e < 16 * Cin; e += 32) {
    int row = e / Cin;
    int c   = e - row * Cin;
    float ctr = (float)fb[(size_t)nrow[row]  * FS + c];
    float nbv = (float)fb[(size_t)nbrow[row] * FS + c];
    A[row * KP + c]        = (__bf16)(nbv - ctr);
    A[row * KP + Cin + c]  = (__bf16)ctr;
  }
  const int pad = KP - C2;
  for (int e = lane; e < 16 * pad; e += 32) {
    int row = e / pad;
    int c   = e - row * pad;
    A[row * KP + C2 + c] = (__bf16)0.0f;
  }
  __syncthreads();

  const int O16 = O >> 4;
  for (int ot = 0; ot < O16; ++ot) {
    const int o = ot * 16 + nc;
    const __bf16* wrow = Wp + (size_t)o * KP;
    __builtin_prefetch(wrow, 0, 1);

    v8f acc;
#pragma unroll
    for (int i = 0; i < 8; ++i) acc[i] = 0.0f;

    for (int kk = 0; kk < KP; kk += 32) {
      v16bf af = load_frag_a(A + nc * KP + kk, half);
      v16bf bf = load_frag_b(wrow + kk, half);
      acc = __builtin_amdgcn_wmma_f32_16x16x32_bf16(false, af, false, bf,
                                                    (short)0, acc, false, false);
    }

    if (PASS == 1) {
      float s1 = 0.0f, s2 = 0.0f;
#pragma unroll
      for (int i = 0; i < 8; ++i) { float v = acc[i]; s1 += v; s2 += v * v; }
      s1 += __shfl_xor(s1, 16);
      s2 += __shfl_xor(s2, 16);
      if (lane < 16) {
        atomicAdd(&sumb[ot * 16 + lane], s1);
        atomicAdd(&sqb [ot * 16 + lane], s2);
      }
    } else {
      const float sc = scaleb[o], sh = shiftb[o];
#pragma unroll
      for (int r = 0; r < 8; ++r) {
        float v = tanhf(acc[r] * sc + sh);
        int m = mtile * 16 + r + 8 * half;        // C/D layout: M = r + 8*(lane>>4)
        int n = m / KNN_K;
        atomic_max_f32(&fbuf[((size_t)b * NPTS + n) * O + o], v);
      }
    }
  }
}

// ---------------------------------------------------------------------------
// Final MLP GEMM: (B*N, 512) x (512, 1024), A rows straight from global fcat.
// PASS==1: stats.  PASS==2: BN+tanh, atomic max + mean-sum into d_out.
// ---------------------------------------------------------------------------
template<int PASS>
__global__ __launch_bounds__(32)
void mlp_gemm_kernel(const __bf16* __restrict__ cat, const __bf16* __restrict__ Wp,
                     float* __restrict__ sumb, float* __restrict__ sqb,
                     const float* __restrict__ scaleb, const float* __restrict__ shiftb,
                     float* __restrict__ out) {
  const int lane  = threadIdx.x;
  const int nc    = lane & 15;
  const int half  = lane >> 4;
  const int mtile = blockIdx.x;                 // (B*N)/16 tiles
  const int ot    = blockIdx.y;                 // 1024/16 output tiles
  const int KP    = CATCH;

  const int o = ot * 16 + nc;
  const __bf16* arow = cat + (size_t)(mtile * 16 + nc) * KP;
  const __bf16* wrow = Wp + (size_t)o * KP;
  __builtin_prefetch(wrow, 0, 1);

  v8f acc;
#pragma unroll
  for (int i = 0; i < 8; ++i) acc[i] = 0.0f;

  for (int kk = 0; kk < KP; kk += 32) {
    v16bf af = load_frag_a(arow + kk, half);
    v16bf bf = load_frag_b(wrow + kk, half);
    acc = __builtin_amdgcn_wmma_f32_16x16x32_bf16(false, af, false, bf,
                                                  (short)0, acc, false, false);
  }

  if (PASS == 1) {
    float s1 = 0.0f, s2 = 0.0f;
#pragma unroll
    for (int i = 0; i < 8; ++i) { float v = acc[i]; s1 += v; s2 += v * v; }
    s1 += __shfl_xor(s1, 16);
    s2 += __shfl_xor(s2, 16);
    if (lane < 16) {
      atomicAdd(&sumb[ot * 16 + lane], s1);
      atomicAdd(&sqb [ot * 16 + lane], s2);
    }
  } else {
    const float sc = scaleb[o], sh = shiftb[o];
#pragma unroll
    for (int r = 0; r < 8; ++r) {
      float v = tanhf(acc[r] * sc + sh);
      int row = mtile * 16 + r + 8 * half;
      int b = row >> 11;                        // N = 2048
      atomic_max_f32(&out[b * 2048 + o], v);
      atomicAdd(&out[b * 2048 + 1024 + o], v);
    }
  }
}

// ---------------------------------------------------------------------------
// launch
// ---------------------------------------------------------------------------
extern "C" void kernel_launch(void* const* d_in, const int* in_sizes, int n_in,
                              void* d_out, int out_size, void* d_ws, size_t ws_size,
                              hipStream_t stream) {
  (void)in_sizes; (void)n_in; (void)out_size; (void)ws_size;
  const float* x    = (const float*)d_in[0];
  const float* feat = (const float*)d_in[1];
  const float* w1 = (const float*)d_in[2];  const float* g1 = (const float*)d_in[3];  const float* b1 = (const float*)d_in[4];
  const float* w2 = (const float*)d_in[5];  const float* g2 = (const float*)d_in[6];  const float* b2 = (const float*)d_in[7];
  const float* w3 = (const float*)d_in[8];  const float* g3 = (const float*)d_in[9];  const float* b3 = (const float*)d_in[10];
  const float* w4 = (const float*)d_in[11]; const float* g4 = (const float*)d_in[12]; const float* b4 = (const float*)d_in[13];
  const float* w5 = (const float*)d_in[14]; const float* g5 = (const float*)d_in[15]; const float* b5 = (const float*)d_in[16];
  float* out = (float*)d_out;

  char* ws = (char*)d_ws;
  size_t off = 0;
  auto alloc = [&](size_t bytes) -> char* {
    char* p = ws + off;
    off = (off + bytes + 255) & ~(size_t)255;
    return p;
  };
  int*    nbr   = (int*)   alloc((size_t)NBATCH * NPTS * KNN_K * 4);
  __bf16* feat0 = (__bf16*)alloc((size_t)NBATCH * NPTS * 3 * 2);
  __bf16* fcat  = (__bf16*)alloc((size_t)NBATCH * NPTS * CATCH * 2);
  float*  fbuf  = (float*) alloc((size_t)NBATCH * NPTS * 256 * 4);
  __bf16* w1p   = (__bf16*)alloc((size_t)64 * 32 * 2);
  __bf16* w2p   = (__bf16*)alloc((size_t)64 * 128 * 2);
  __bf16* w3p   = (__bf16*)alloc((size_t)128 * 128 * 2);
  __bf16* w4p   = (__bf16*)alloc((size_t)256 * 256 * 2);
  __bf16* w5p   = (__bf16*)alloc((size_t)1024 * 512 * 2);
  float*  sumb  = (float*) alloc(1024 * 4);
  float*  sqb   = (float*) alloc(1024 * 4);
  float*  scb   = (float*) alloc(1024 * 4);
  float*  shb   = (float*) alloc(1024 * 4);

  // kNN once (same x for every block)
  knn_kernel<<<dim3(NPTS / 256, NBATCH), 256, 0, stream>>>(x, nbr);

  // input features -> bf16
  {
    int n = NBATCH * NPTS * 3;
    cvt_feat_kernel<<<(n + 255) / 256, 256, 0, stream>>>(feat, feat0, n);
  }
  // pad/convert weights
  cvt_w_kernel<<<(64 * 32   + 255) / 256, 256, 0, stream>>>(w1, w1p, 64,   6,   32);
  cvt_w_kernel<<<(64 * 128  + 255) / 256, 256, 0, stream>>>(w2, w2p, 64,   128, 128);
  cvt_w_kernel<<<(128 * 128 + 255) / 256, 256, 0, stream>>>(w3, w3p, 128,  128, 128);
  cvt_w_kernel<<<(256 * 256 + 255) / 256, 256, 0, stream>>>(w4, w4p, 256,  256, 256);
  cvt_w_kernel<<<((1024 * 512) + 255) / 256, 256, 0, stream>>>(w5, w5p, 1024, 512, 512);

  const float edge_cnt = (float)NBATCH * NPTS * KNN_K;  // 327680
  const dim3  egrid((NPTS * KNN_K) / 16, NBATCH);

  auto run_block = [&](const __bf16* fin, int FS, int FOFF, int Cin, int KP, int O,
                       const __bf16* wp, const float* g, const float* bb, int coff) {
    fill_kernel<<<(O + 255) / 256, 256, 0, stream>>>(sumb, 0.0f, O);
    fill_kernel<<<(O + 255) / 256, 256, 0, stream>>>(sqb,  0.0f, O);
    size_t shbytes = 128 + (size_t)16 * KP * 2;
    edge_gemm_kernel<1><<<egrid, 32, shbytes, stream>>>(
        fin, FS, FOFF, Cin, nbr, wp, O, KP, sumb, sqb, nullptr, nullptr, nullptr);
    finalize_kernel<<<(O + 255) / 256, 256, 0, stream>>>(sumb, sqb, g, bb, scb, shb, O, edge_cnt);
    int nf = NBATCH * NPTS * O;
    fill_kernel<<<(nf + 255) / 256, 256, 0, stream>>>(fbuf, -2.0f, nf);
    edge_gemm_kernel<2><<<egrid, 32, shbytes, stream>>>(
        fin, FS, FOFF, Cin, nbr, wp, O, KP, nullptr, nullptr, scb, shb, fbuf);
    cvt_cat_kernel<<<(nf + 255) / 256, 256, 0, stream>>>(fbuf, fcat, O, coff);
  };

  run_block(feat0, 3,     0,   3,   32,  64,  w1p, g1, b1, 0);
  run_block(fcat,  CATCH, 0,   64,  128, 64,  w2p, g2, b2, 64);
  run_block(fcat,  CATCH, 64,  64,  128, 128, w3p, g3, b3, 128);
  run_block(fcat,  CATCH, 128, 128, 256, 256, w4p, g4, b4, 256);

  // final MLP + BN + tanh + max/mean pooling
  fill_kernel<<<4, 256, 0, stream>>>(sumb, 0.0f, 1024);
  fill_kernel<<<4, 256, 0, stream>>>(sqb,  0.0f, 1024);
  const dim3 mgrid((NBATCH * NPTS) / 16, 1024 / 16);
  mlp_gemm_kernel<1><<<mgrid, 32, 0, stream>>>(fcat, w5p, sumb, sqb, nullptr, nullptr, nullptr);
  finalize_kernel<<<4, 256, 0, stream>>>(sumb, sqb, g5, b5, scb, shb, 1024, (float)NBATCH * NPTS);
  init_out_kernel<<<(NBATCH * 2048 + 255) / 256, 256, 0, stream>>>(out);
  mlp_gemm_kernel<2><<<mgrid, 32, 0, stream>>>(fcat, w5p, nullptr, nullptr, scb, shb, out);
  finish_mean_kernel<<<(NBATCH * 1024 + 255) / 256, 256, 0, stream>>>(out);
}
